// MLABmmLayerFP8_29454885716349
// MI455X (gfx1250) — 8-bit matrix operand order repaired and checked by reading, compiled, not run on hardware
//
#include <hip/hip_runtime.h>
#include <hip/hip_bf16.h>

typedef int   v16i __attribute__((ext_vector_type(16)));
typedef float v8f  __attribute__((ext_vector_type(8)));

namespace {
constexpr int kT = 4096;
constexpr int kH = 128;
constexpr int kD = 128;
constexpr int kL = 512;

constexpr int BT = 128;
constexpr int BL = 128;
constexpr int LDS_STRIDE = 144;
}

__device__ __forceinline__ float bf16_as_f32(unsigned short u) {
  union { unsigned u; float f; } c;
  c.u = (unsigned)u << 16;
  return c.f;
}

__device__ __forceinline__ unsigned pack4_fp8(unsigned ab, unsigned cd) {
  float f0 = bf16_as_f32((unsigned short)(ab & 0xffffu));
  float f1 = bf16_as_f32((unsigned short)(ab >> 16));
  float f2 = bf16_as_f32((unsigned short)(cd & 0xffffu));
  float f3 = bf16_as_f32((unsigned short)(cd >> 16));
  int r = 0;
  r = __builtin_amdgcn_cvt_pk_fp8_f32(f0, f1, r, false);
  r = __builtin_amdgcn_cvt_pk_fp8_f32(f2, f3, r, true);
  return (unsigned)r;
}

__global__ __launch_bounds__(256)
void mla_bmm_fp8_wmma(const __hip_bfloat16* __restrict__ x,
                      const __hip_bfloat16* __restrict__ w,
                      __hip_bfloat16* __restrict__ out) {
  __shared__ __align__(16) unsigned char xs[BT * LDS_STRIDE];
  __shared__ __align__(16) unsigned char wsb[BL * LDS_STRIDE];

  const int tBase = blockIdx.x * BT;
  const int lBase = blockIdx.y * BL;
  const int h     = blockIdx.z;

  const int tid  = threadIdx.x;
  const int row  = tid >> 1;
  const int half = tid & 1;

  {
    const uint4* g = reinterpret_cast<const uint4*>(
        x + ((size_t)(tBase + row) * kH + h) * kD + half * 64);
    uint4* s = reinterpret_cast<uint4*>(xs + row * LDS_STRIDE + half * 64);
#pragma unroll
    for (int i = 0; i < 4; ++i) {
      uint4 q0 = g[2 * i + 0];
      uint4 q1 = g[2 * i + 1];
      uint4 o;
      o.x = pack4_fp8(q0.x, q0.y);
      o.y = pack4_fp8(q0.z, q0.w);
      o.z = pack4_fp8(q1.x, q1.y);
      o.w = pack4_fp8(q1.z, q1.w);
      s[i] = o;
    }
  }
  {
    const uint4* g = reinterpret_cast<const uint4*>(
        w + ((size_t)h * kL + lBase + row) * kD + half * 64);
    uint4* s = reinterpret_cast<uint4*>(wsb + row * LDS_STRIDE + half * 64);
#pragma unroll
    for (int i = 0; i < 4; ++i) {
      uint4 q0 = g[2 * i + 0];
      uint4 q1 = g[2 * i + 1];
      uint4 o;
      o.x = pack4_fp8(q0.x, q0.y);
      o.y = pack4_fp8(q0.z, q0.w);
      o.z = pack4_fp8(q1.x, q1.y);
      o.w = pack4_fp8(q1.z, q1.w);
      s[i] = o;
    }
  }
  __syncthreads();

  const int wave   = tid >> 5;
  const int lane   = tid & 31;
  const int laneM  = lane & 15;
  const int laneHi = lane >> 4;

  v16i A;
  {
    const unsigned char* arow = xs + (wave * 16 + laneM) * LDS_STRIDE;
    const int kh = laneHi * 16;
#pragma unroll
    for (int p = 0; p < 8; ++p) {
      const int off = ((p >> 1) * 32) + ((p & 1) * 8) + kh;
      uint2 v = *reinterpret_cast<const uint2*>(arow + off);
      A[2 * p + 0] = (int)v.x;
      A[2 * p + 1] = (int)v.y;
    }
  }

  v8f acc[8];
  const v8f vzero = {};
#pragma unroll
  for (int n = 0; n < 8; ++n) acc[n] = vzero;

#pragma unroll
  for (int n = 0; n < 8; ++n) {
    v16i B;
    const unsigned char* brow = wsb + (n * 16 + laneM) * LDS_STRIDE;
    const int kb = laneHi * 16;
#pragma unroll
    for (int gq = 0; gq < 4; ++gq) {
      uint4 v = *reinterpret_cast<const uint4*>(brow + gq * 32 + kb);
      B[4 * gq + 0] = (int)v.x;
      B[4 * gq + 1] = (int)v.y;
      B[4 * gq + 2] = (int)v.z;
      B[4 * gq + 3] = (int)v.w;
    }
    acc[n] = __builtin_amdgcn_wmma_f32_16x16x128_fp8_fp8(A, B, (short)0, acc[n],
                                                         false, false);
  }

#pragma unroll
  for (int n = 0; n < 8; ++n) {
#pragma unroll
    for (int r = 0; r < 8; ++r) {
      const int t = tBase + wave * 16 + laneHi * 8 + r;
      const int l = lBase + n * 16 + laneM;
      __hip_bfloat16 hb = __float2bfloat16(acc[n][r]);
      unsigned short bits;
      __builtin_memcpy(&bits, &hb, sizeof(bits));
      unsigned short* dst = reinterpret_cast<unsigned short*>(out) +
                            (((size_t)t * kH + h) * kL + l);
      __builtin_nontemporal_store(bits, dst);
    }
  }
}

extern "C" void kernel_launch(void* const* d_in, const int* in_sizes, int n_in,
                              void* d_out, int out_size, void* d_ws, size_t ws_size,
                              hipStream_t stream) {
  (void)in_sizes; (void)n_in; (void)out_size; (void)d_ws; (void)ws_size;
  const __hip_bfloat16* x = (const __hip_bfloat16*)d_in[0];
  const __hip_bfloat16* w = (const __hip_bfloat16*)d_in[1];
  __hip_bfloat16* out = (__hip_bfloat16*)d_out;

  dim3 grid(kT / BT, kL / BL, kH);
  mla_bmm_fp8_wmma<<<grid, dim3(256), 0, stream>>>(x, w, out);
}
